// MixedAttention_72602127172196
// MI455X (gfx1250) — matrix operand order repaired and checked by reading, not compiled, not run on hardware
//
#include <hip/hip_runtime.h>
#include <cstdint>

// ---------------------------------------------------------------------------
// ConvBERT-style mixed attention for MI455X (gfx1250), wave32 + WMMA bf16.
// B=4, S=2048, HIDDEN=768, N_HEADS=6, HEAD_DIM=64, ALL_HEAD=384, K=9
// ---------------------------------------------------------------------------

#define BATCH    4
#define SEQ      2048
#define ROWS     (BATCH * SEQ)      // 8192
#define HID      768
#define NH       6
#define HD       64
#define AH       (NH * HD)          // 384
#define KW       9

typedef __attribute__((ext_vector_type(8)))  float          f32x8;
typedef __attribute__((ext_vector_type(16))) __bf16         bf16x16;
typedef __attribute__((ext_vector_type(8)))  unsigned short ush8;
typedef __attribute__((ext_vector_type(4)))  int            i32x4;

#define WMMA_BF16(a, b, c) \
  __builtin_amdgcn_wmma_f32_16x16x32_bf16(false, (a), false, (b), (short)0, (c), false, false)

// gfx1250 async global->LDS copy path (guarded; falls back to load+store).
// Signature (verified on this toolchain): (int4 AS1*, int4 AS3*, imm, imm)
#if defined(__has_builtin)
#if __has_builtin(__builtin_amdgcn_global_load_async_to_lds_b128) && \
    __has_builtin(__builtin_amdgcn_s_wait_asynccnt)
#define USE_ASYNC_LDS 1
#endif
#endif
#ifndef USE_ASYNC_LDS
#define USE_ASYNC_LDS 0
#endif

#if USE_ASYNC_LDS
typedef __attribute__((address_space(1))) i32x4 gi32x4;   // global int4
typedef __attribute__((address_space(3))) i32x4 li32x4;   // LDS int4
#define ASYNC_COPY_B128(src, dst)                                            \
  __builtin_amdgcn_global_load_async_to_lds_b128(                            \
      (gi32x4*)(uintptr_t)(src), (li32x4*)(unsigned)(uintptr_t)(dst), 0, 0)
#define ASYNC_WAIT(n) __builtin_amdgcn_s_wait_asynccnt(n)
#else
#define ASYNC_COPY_B128(src, dst) \
  (*reinterpret_cast<ush8*>(dst) = *reinterpret_cast<const ush8*>(src))
#define ASYNC_WAIT(n)
#endif

// fp32 -> bf16 round-to-nearest-even
__device__ __forceinline__ unsigned short f2bf(float f) {
  unsigned int u = __float_as_uint(f);
  u += 0x7FFFu + ((u >> 16) & 1u);
  return (unsigned short)(u >> 16);
}

// A-matrix 16x32 bf16 tile (ISA 7.12.2): lanes 0-15 row M=lane hold K 0..7 /
// 16..23, lanes 16-31 same rows hold K 8..15 / 24..31.
__device__ __forceinline__ bf16x16 load_a_tile(const unsigned short* base, int stride, int lane) {
  const int r = lane & 15;
  const int c = (lane >> 4) << 3;               // 0 or 8 halves
  ush8 lo = *reinterpret_cast<const ush8*>(base + (size_t)r * stride + c);
  ush8 hi = *reinterpret_cast<const ush8*>(base + (size_t)r * stride + c + 16);
  union { ush8 u[2]; bf16x16 v; } cvt;
  cvt.u[0] = lo; cvt.u[1] = hi;
  return cvt.v;
}

// B-matrix 32x16 bf16 tile: memory holds [N][Kdim] row-major; lanes 0-15
// (N = lane) hold K 0..7 / 16..23, lanes 16-31 hold K 8..15 / 24..31.
__device__ __forceinline__ bf16x16 load_b_tile(const unsigned short* base, int stride, int lane) {
  const int r = lane & 15;
  const int c = (lane >> 4) << 3;               // 0 or 8 halves
  ush8 lo = *reinterpret_cast<const ush8*>(base + (size_t)r * stride + c);
  ush8 hi = *reinterpret_cast<const ush8*>(base + (size_t)r * stride + c + 16);
  union { ush8 u[2]; bf16x16 v; } cvt;
  cvt.u[0] = lo; cvt.u[1] = hi;
  return cvt.v;
}

__device__ __forceinline__ float redmax16(float v) {
  v = fmaxf(v, __shfl_xor(v, 1, 32));
  v = fmaxf(v, __shfl_xor(v, 2, 32));
  v = fmaxf(v, __shfl_xor(v, 4, 32));
  v = fmaxf(v, __shfl_xor(v, 8, 32));
  return v;
}
__device__ __forceinline__ float redsum16(float v) {
  v += __shfl_xor(v, 1, 32);
  v += __shfl_xor(v, 2, 32);
  v += __shfl_xor(v, 4, 32);
  v += __shfl_xor(v, 8, 32);
  return v;
}

// --------------------------- conversion kernels ----------------------------

__global__ __launch_bounds__(256) void k_cvt_x(const float* __restrict__ x,
                                               unsigned short* __restrict__ xb) {
  int i = blockIdx.x * 256 + threadIdx.x;       // ROWS*HID = 6291456 exact
  xb[i] = f2bf(x[i]);
}

__global__ __launch_bounds__(256) void k_cvt_w(
    const float* __restrict__ wq, const float* __restrict__ wk,
    const float* __restrict__ wv, const float* __restrict__ wco,
    const float* __restrict__ wpw,
    unsigned short* __restrict__ oq, unsigned short* __restrict__ ok,
    unsigned short* __restrict__ ov, unsigned short* __restrict__ oco,
    unsigned short* __restrict__ opw) {
  const float* s; unsigned short* d;
  switch (blockIdx.y) {
    case 0:  s = wq;  d = oq;  break;
    case 1:  s = wk;  d = ok;  break;
    case 2:  s = wv;  d = ov;  break;
    case 3:  s = wco; d = oco; break;
    default: s = wpw; d = opw; break;
  }
  int i = blockIdx.x * 256 + threadIdx.x;       // AH*HID = 294912 exact
  d[i] = f2bf(s[i]);
}

// Depthwise conv over seq (groups=HID, K=9, same padding), output bf16 [row][c]
__global__ __launch_bounds__(256) void k_dwconv(const float* __restrict__ x,
                                                const float* __restrict__ dw,
                                                unsigned short* __restrict__ o) {
  int row = blockIdx.x;
  int c   = blockIdx.y * 256 + threadIdx.x;     // 0..767
  int b = row >> 11, s = row & (SEQ - 1);
  float acc = 0.f;
#pragma unroll
  for (int k = 0; k < KW; ++k) {
    int sp = s + k - KW / 2;
    if (sp >= 0 && sp < SEQ)
      acc += x[((size_t)(b * SEQ + sp)) * HID + c] * dw[c * KW + k];
  }
  o[(size_t)row * HID + c] = f2bf(acc);
}

// q,k,v fp32 [row][h*64+d] -> bf16 per-head layouts: qb/kb [b,h,s,d] (q scaled
// by 1/sqrt(64)), vT [b,h,d,s] so PV B-tiles are contiguous per lane.
__global__ __launch_bounds__(128) void k_cvt_qkv(
    const float* __restrict__ q, const float* __restrict__ k,
    const float* __restrict__ v, unsigned short* __restrict__ qb,
    unsigned short* __restrict__ kb, unsigned short* __restrict__ vt) {
  int row = blockIdx.x;
  int col = blockIdx.y * 128 + threadIdx.x;     // 0..383
  int b = row >> 11, s = row & (SEQ - 1);
  int h = col >> 6, d = col & 63;
  size_t src = (size_t)row * AH + col;
  size_t bh  = (size_t)(b * NH + h);
  qb[(bh * SEQ + s) * HD + d] = f2bf(q[src] * 0.125f);
  kb[(bh * SEQ + s) * HD + d] = f2bf(k[src]);
  vt[(bh * HD + d) * SEQ + s] = f2bf(v[src]);
}

// ------------------------------ WMMA GEMM ----------------------------------
// Y[M=8192, N=384] = A_bf16[8192,768] @ W_bf16[384,768]^T + bias
// blockIdx.z picks {q, k, v, co, kc}. Block = 4 waves; each wave computes a
// 32x64 tile. The 64x32 B panel (shared by all waves) is staged into LDS with
// async global->LDS copies (double buffered: Bs[0]=even 32-panels, Bs[1]=odd).
// K-loop is unrolled by 2 with ping-pong A-fragment register sets so there is
// no loop-carried register rotation (no mov chains / hazard nops).
#define BSTRIDE 40   // halves per LDS B row (80B): 16-lane b128 reads conflict-free

__global__ __launch_bounds__(128) void k_gemm(
    const unsigned short* __restrict__ xb, const unsigned short* __restrict__ dwb,
    const unsigned short* __restrict__ wq, const unsigned short* __restrict__ wk,
    const unsigned short* __restrict__ wv, const unsigned short* __restrict__ wco,
    const unsigned short* __restrict__ wpw,
    const float* __restrict__ bq, const float* __restrict__ bk,
    const float* __restrict__ bv, const float* __restrict__ bco,
    const float* __restrict__ bcv,
    float* __restrict__ q, float* __restrict__ k, float* __restrict__ v,
    float* __restrict__ co, float* __restrict__ kc) {
  const unsigned short* A; const unsigned short* W; const float* bias; float* Y;
  switch (blockIdx.z) {
    case 0:  A = xb;  W = wq;  bias = bq;  Y = q;  break;
    case 1:  A = xb;  W = wk;  bias = bk;  Y = k;  break;
    case 2:  A = xb;  W = wv;  bias = bv;  Y = v;  break;
    case 3:  A = xb;  W = wco; bias = bco; Y = co; break;
    default: A = dwb; W = wpw; bias = bcv; Y = kc; break;
  }
  __shared__ unsigned short Bs[2][64 * BSTRIDE];

  const int t    = threadIdx.x;
  const int lane = t & 31;
  const int wave = t >> 5;
  const int m0 = blockIdx.x * 128 + wave * 32;   // wave owns rows m0..m0+31
  const int n0 = blockIdx.y * 64;
  const unsigned short* A0 = A + (size_t)(m0 +  0) * HID;
  const unsigned short* A1 = A + (size_t)(m0 + 16) * HID;

  // stage one 64x32 B panel (4KB) into Bs[buf]: 256 16B chunks, 2 per thread
  auto stage_B = [&](int buf, int kk) {
#pragma unroll
    for (int i = 0; i < 2; ++i) {
      int chunk = t + i * 128;                   // 0..255
      int row = chunk >> 2;                      // 0..63
      int qd  = chunk & 3;                       // 16B quarter of the row
      const unsigned short* src = W + (size_t)(n0 + row) * HID + kk + qd * 8;
      unsigned short* dst = &Bs[buf][row * BSTRIDE + qd * 8];
      ASYNC_COPY_B128(src, dst);
    }
  };

  f32x8 c00 = {}, c01 = {}, c02 = {}, c03 = {};  // rows m0..m0+15
  f32x8 c10 = {}, c11 = {}, c12 = {}, c13 = {};  // rows m0+16..m0+31

  // 8 WMMAs against one LDS panel using one A-fragment pair
  auto gemm_step = [&](const unsigned short* bs, bf16x16 a0, bf16x16 a1) {
    bf16x16 b0 = load_b_tile(bs +  0 * BSTRIDE, BSTRIDE, lane);
    bf16x16 b1 = load_b_tile(bs + 16 * BSTRIDE, BSTRIDE, lane);
    bf16x16 b2 = load_b_tile(bs + 32 * BSTRIDE, BSTRIDE, lane);
    bf16x16 b3 = load_b_tile(bs + 48 * BSTRIDE, BSTRIDE, lane);
    c00 = WMMA_BF16(a0, b0, c00);
    c01 = WMMA_BF16(a0, b1, c01);
    c02 = WMMA_BF16(a0, b2, c02);
    c03 = WMMA_BF16(a0, b3, c03);
    c10 = WMMA_BF16(a1, b0, c10);
    c11 = WMMA_BF16(a1, b1, c11);
    c12 = WMMA_BF16(a1, b2, c12);
    c13 = WMMA_BF16(a1, b3, c13);
  };

  stage_B(0, 0);
  bf16x16 aA0 = load_a_tile(A0, HID, lane);
  bf16x16 aA1 = load_a_tile(A1, HID, lane);
  bf16x16 aB0, aB1;

#pragma unroll 1
  for (int kk = 0; kk < HID; kk += 64) {
    // ---- sub-step 0: consume Bs[0]/aA*, prefetch Bs[1]/aB* (kk+32) ----
    stage_B(1, kk + 32);                         // kk+32 < HID always here
    ASYNC_WAIT(2);                               // panel kk resident (this thread)
    __syncthreads();                             // ... and for all waves
    aB0 = load_a_tile(A0 + kk + 32, HID, lane);
    aB1 = load_a_tile(A1 + kk + 32, HID, lane);
    gemm_step(&Bs[0][0], aA0, aA1);
    __syncthreads();                             // Bs[0] reads done before restage

    // ---- sub-step 1: consume Bs[1]/aB*, prefetch Bs[0]/aA* (kk+64) ----
    if (kk + 64 < HID) {
      stage_B(0, kk + 64);
      ASYNC_WAIT(2);
    } else {
      ASYNC_WAIT(0);
    }
    __syncthreads();
    if (kk + 64 < HID) {
      aA0 = load_a_tile(A0 + kk + 64, HID, lane);
      aA1 = load_a_tile(A1 + kk + 64, HID, lane);
    }
    gemm_step(&Bs[1][0], aB0, aB1);
    __syncthreads();
  }

  const int nn = lane & 15;
  const int mo = (lane >> 4) << 3;
  const float e0 = bias[n0 +  0 + nn];
  const float e1 = bias[n0 + 16 + nn];
  const float e2 = bias[n0 + 32 + nn];
  const float e3 = bias[n0 + 48 + nn];
#pragma unroll
  for (int r = 0; r < 8; ++r) {
    size_t o = (size_t)(m0 + mo + r) * AH + n0;
    Y[o +  0 + nn] = c00[r] + e0;
    Y[o + 16 + nn] = c01[r] + e1;
    Y[o + 32 + nn] = c02[r] + e2;
    Y[o + 48 + nn] = c03[r] + e3;
    o += (size_t)16 * AH;
    Y[o +  0 + nn] = c10[r] + e0;
    Y[o + 16 + nn] = c11[r] + e1;
    Y[o + 32 + nn] = c12[r] + e2;
    Y[o + 48 + nn] = c13[r] + e3;
  }
}

// --------------------- span-conv kernel layer + softmax --------------------
__global__ __launch_bounds__(64) void k_ckl(
    const float* __restrict__ q, const float* __restrict__ kc,
    const float* __restrict__ Wck, const float* __restrict__ bck,
    float* __restrict__ ckl) {
  __shared__ float ca[AH];
  __shared__ float sraw[NH * KW];
  const int row = blockIdx.x;
  const int t = threadIdx.x;
  const size_t base = (size_t)row * AH;
  for (int i = t; i < AH; i += 64) ca[i] = kc[base + i] * q[base + i];
  __syncthreads();
  if (t < NH * KW) {
    float acc = bck[t];
    const float* wr = Wck + (size_t)t * AH;
    for (int i = 0; i < AH; ++i) acc += wr[i] * ca[i];
    sraw[t] = acc;
  }
  __syncthreads();
  if (t < NH) {
    float m = -1e30f;
#pragma unroll
    for (int k = 0; k < KW; ++k) m = fmaxf(m, sraw[t * KW + k]);
    float e[KW], s = 0.f;
#pragma unroll
    for (int k = 0; k < KW; ++k) { e[k] = __expf(sraw[t * KW + k] - m); s += e[k]; }
    float inv = 1.f / s;
#pragma unroll
    for (int k = 0; k < KW; ++k) ckl[(size_t)row * (NH * KW) + t * KW + k] = e[k] * inv;
  }
}

// --------------------------- 9-tap dynamic conv ----------------------------
__global__ __launch_bounds__(64) void k_convout(const float* __restrict__ co,
                                                const float* __restrict__ ckl,
                                                float* __restrict__ out) {
  const int row = blockIdx.x, h = blockIdx.y, d = threadIdx.x;
  const int b = row >> 11, s = row & (SEQ - 1);
  float w[KW];
#pragma unroll
  for (int k = 0; k < KW; ++k) w[k] = ckl[(size_t)row * (NH * KW) + h * KW + k];
  float acc = 0.f;
#pragma unroll
  for (int k = 0; k < KW; ++k) {
    int sp = s + k - KW / 2;
    if (sp >= 0 && sp < SEQ)
      acc += co[((size_t)(b * SEQ + sp)) * AH + h * HD + d] * w[k];
  }
  out[(size_t)row * (2 * AH) + AH + h * HD + d] = acc;
}

// --------------------------- flash attention -------------------------------
// grid (16, 24): blockIdx.y = b*6+h, blockIdx.x = 128-row q block, 8 waves,
// each wave owns 16 q rows. 32-key steps with online softmax; outer loop
// unrolled by 2 with ping-pong K-tile register sets (kA*/kB*) so prefetch
// needs no loop-carried copies. V tiles and next K tiles are issued before
// the softmax VALU block, which hides their latency.
#define PSTRIDE 40   // padded P row (halves) -> conflict-free A-layout reload

__global__ __launch_bounds__(256) void k_attn(const unsigned short* __restrict__ qb,
                                              const unsigned short* __restrict__ kb,
                                              const unsigned short* __restrict__ vt,
                                              float* __restrict__ out) {
  __shared__ unsigned short pbuf[8][16 * PSTRIDE];
  const int lane = threadIdx.x & 31;
  const int wave = threadIdx.x >> 5;
  const int bh = blockIdx.y;
  const int b = bh / NH, h = bh % NH;
  const int q0 = blockIdx.x * 128 + wave * 16;
  const int half = lane >> 4, nn = lane & 15;

  const unsigned short* Q  = qb + (size_t)bh * SEQ * HD;
  const unsigned short* Kp = kb + (size_t)bh * SEQ * HD;
  const unsigned short* Vt = vt + (size_t)bh * HD * SEQ;
  unsigned short* P = pbuf[wave];

  const bf16x16 aq0 = load_a_tile(Q + (size_t)q0 * HD +  0, HD, lane);
  const bf16x16 aq1 = load_a_tile(Q + (size_t)q0 * HD + 32, HD, lane);

  float mrow[8], lrow[8];
#pragma unroll
  for (int r = 0; r < 8; ++r) { mrow[r] = -1e30f; lrow[r] = 0.f; }
  f32x8 acc0 = {}, acc1 = {}, acc2 = {}, acc3 = {};

  // online softmax + P@V for one 32-key step (scores in s0/s1, V tiles given)
  auto attn_tail = [&](f32x8 s0, f32x8 s1,
                       bf16x16 v0, bf16x16 v1, bf16x16 v2, bf16x16 v3) {
    float p0[8], p1[8], sc[8];
#pragma unroll
    for (int r = 0; r < 8; ++r) {
      float rm = redmax16(fmaxf(s0[r], s1[r]));
      float mn = fmaxf(mrow[r], rm);
      sc[r] = __expf(mrow[r] - mn);
      p0[r] = __expf(s0[r] - mn);
      p1[r] = __expf(s1[r] - mn);
      float rs = redsum16(p0[r] + p1[r]);
      lrow[r] = lrow[r] * sc[r] + rs;
      mrow[r] = mn;
    }
#pragma unroll
    for (int r = 0; r < 8; ++r) {
      acc0[r] *= sc[r]; acc1[r] *= sc[r]; acc2[r] *= sc[r]; acc3[r] *= sc[r];
    }
#pragma unroll
    for (int r = 0; r < 8; ++r) {
      P[(r + 8 * half) * PSTRIDE + nn]      = f2bf(p0[r]);
      P[(r + 8 * half) * PSTRIDE + 16 + nn] = f2bf(p1[r]);
    }
    __builtin_amdgcn_wave_barrier();            // intra-wave; DS ops are in-order
    bf16x16 pa = load_a_tile(P, PSTRIDE, lane);
    __builtin_amdgcn_wave_barrier();
    acc0 = WMMA_BF16(pa, v0, acc0);
    acc1 = WMMA_BF16(pa, v1, acc1);
    acc2 = WMMA_BF16(pa, v2, acc2);
    acc3 = WMMA_BF16(pa, v3, acc3);
  };

  bf16x16 kA00 = load_b_tile(Kp + (size_t) 0 * HD +  0, HD, lane);
  bf16x16 kA01 = load_b_tile(Kp + (size_t) 0 * HD + 32, HD, lane);
  bf16x16 kA10 = load_b_tile(Kp + (size_t)16 * HD +  0, HD, lane);
  bf16x16 kA11 = load_b_tile(Kp + (size_t)16 * HD + 32, HD, lane);
  bf16x16 kB00, kB01, kB10, kB11;

#pragma unroll 1
  for (int kk = 0; kk < SEQ; kk += 64) {
    // ---- step A: keys kk..kk+31 (uses kA*, prefetches kB* for kk+32) ----
    f32x8 s0 = {}, s1 = {};
    s0 = WMMA_BF16(aq0, kA00, s0);
    s0 = WMMA_BF16(aq1, kA01, s0);
    s1 = WMMA_BF16(aq0, kA10, s1);
    s1 = WMMA_BF16(aq1, kA11, s1);
    bf16x16 v0 = load_b_tile(Vt + (size_t)( 0) * SEQ + kk, SEQ, lane);
    bf16x16 v1 = load_b_tile(Vt + (size_t)(16) * SEQ + kk, SEQ, lane);
    bf16x16 v2 = load_b_tile(Vt + (size_t)(32) * SEQ + kk, SEQ, lane);
    bf16x16 v3 = load_b_tile(Vt + (size_t)(48) * SEQ + kk, SEQ, lane);
    kB00 = load_b_tile(Kp + (size_t)(kk + 32) * HD +  0, HD, lane);
    kB01 = load_b_tile(Kp + (size_t)(kk + 32) * HD + 32, HD, lane);
    kB10 = load_b_tile(Kp + (size_t)(kk + 48) * HD +  0, HD, lane);
    kB11 = load_b_tile(Kp + (size_t)(kk + 48) * HD + 32, HD, lane);
    attn_tail(s0, s1, v0, v1, v2, v3);

    // ---- step B: keys kk+32..kk+63 (uses kB*, prefetches kA* for kk+64) ----
    f32x8 t0 = {}, t1 = {};
    t0 = WMMA_BF16(aq0, kB00, t0);
    t0 = WMMA_BF16(aq1, kB01, t0);
    t1 = WMMA_BF16(aq0, kB10, t1);
    t1 = WMMA_BF16(aq1, kB11, t1);
    v0 = load_b_tile(Vt + (size_t)( 0) * SEQ + kk + 32, SEQ, lane);
    v1 = load_b_tile(Vt + (size_t)(16) * SEQ + kk + 32, SEQ, lane);
    v2 = load_b_tile(Vt + (size_t)(32) * SEQ + kk + 32, SEQ, lane);
    v3 = load_b_tile(Vt + (size_t)(48) * SEQ + kk + 32, SEQ, lane);
    if (kk + 64 < SEQ) {
      kA00 = load_b_tile(Kp + (size_t)(kk + 64) * HD +  0, HD, lane);
      kA01 = load_b_tile(Kp + (size_t)(kk + 64) * HD + 32, HD, lane);
      kA10 = load_b_tile(Kp + (size_t)(kk + 80) * HD +  0, HD, lane);
      kA11 = load_b_tile(Kp + (size_t)(kk + 80) * HD + 32, HD, lane);
    }
    attn_tail(t0, t1, v0, v1, v2, v3);
  }

#pragma unroll
  for (int r = 0; r < 8; ++r) {
    float inv = 1.0f / lrow[r];
    int row = q0 + r + 8 * half;
    size_t o = ((size_t)(b * SEQ + row)) * (2 * AH) + h * HD;
    out[o +  0 + nn] = acc0[r] * inv;
    out[o + 16 + nn] = acc1[r] * inv;
    out[o + 32 + nn] = acc2[r] * inv;
    out[o + 48 + nn] = acc3[r] * inv;
  }
}

// ------------------------------- launcher ----------------------------------

extern "C" void kernel_launch(void* const* d_in, const int* in_sizes, int n_in,
                              void* d_out, int out_size, void* d_ws, size_t ws_size,
                              hipStream_t stream) {
  (void)in_sizes; (void)n_in; (void)out_size; (void)ws_size;
  const float* x         = (const float*)d_in[0];
  const float* Wq        = (const float*)d_in[1];
  const float* bq        = (const float*)d_in[2];
  const float* Wk        = (const float*)d_in[3];
  const float* bk        = (const float*)d_in[4];
  const float* Wv        = (const float*)d_in[5];
  const float* bv        = (const float*)d_in[6];
  const float* dw        = (const float*)d_in[7];
  const float* pw        = (const float*)d_in[8];
  const float* conv_bias = (const float*)d_in[9];
  const float* Wck       = (const float*)d_in[10];
  const float* bck       = (const float*)d_in[11];
  const float* Wco       = (const float*)d_in[12];
  const float* bco       = (const float*)d_in[13];
  float* out = (float*)d_out;

  char* ws = (char*)d_ws;
  auto carve = [&ws](size_t bytes) {
    char* p = ws;
    ws += (bytes + 255) & ~(size_t)255;
    return p;
  };
  unsigned short* xb   = (unsigned short*)carve((size_t)ROWS * HID * 2);
  unsigned short* wqb  = (unsigned short*)carve((size_t)AH * HID * 2);
  unsigned short* wkb  = (unsigned short*)carve((size_t)AH * HID * 2);
  unsigned short* wvb  = (unsigned short*)carve((size_t)AH * HID * 2);
  unsigned short* wcob = (unsigned short*)carve((size_t)AH * HID * 2);
  unsigned short* wpwb = (unsigned short*)carve((size_t)AH * HID * 2);
  unsigned short* dwb  = (unsigned short*)carve((size_t)ROWS * HID * 2);
  float* q   = (float*)carve((size_t)ROWS * AH * 4);
  float* k   = (float*)carve((size_t)ROWS * AH * 4);
  float* v   = (float*)carve((size_t)ROWS * AH * 4);
  float* co  = (float*)carve((size_t)ROWS * AH * 4);
  float* kc  = (float*)carve((size_t)ROWS * AH * 4);
  unsigned short* qbh = (unsigned short*)carve((size_t)ROWS * AH * 2);
  unsigned short* kbh = (unsigned short*)carve((size_t)ROWS * AH * 2);
  unsigned short* vth = (unsigned short*)carve((size_t)ROWS * AH * 2);
  float* ckl = (float*)carve((size_t)ROWS * NH * KW * 4);

  // 1) precision conversion + depthwise conv (independent)
  k_cvt_x<<<dim3(ROWS * HID / 256), dim3(256), 0, stream>>>(x, xb);
  k_cvt_w<<<dim3(AH * HID / 256, 5), dim3(256), 0, stream>>>(
      Wq, Wk, Wv, Wco, pw, wqb, wkb, wvb, wcob, wpwb);
  k_dwconv<<<dim3(ROWS, HID / 256), dim3(256), 0, stream>>>(x, dw, dwb);

  // 2) five WMMA GEMMs: q, k, v, conv_out_layer, pointwise(kc)
  k_gemm<<<dim3(ROWS / 128, AH / 64, 5), dim3(128), 0, stream>>>(
      xb, dwb, wqb, wkb, wvb, wcob, wpwb,
      bq, bk, bv, bco, conv_bias, q, k, v, co, kc);

  // 3) head-layout bf16 repack (+ q scale), span-conv kernel softmax
  k_cvt_qkv<<<dim3(ROWS, AH / 128), dim3(128), 0, stream>>>(q, k, v, qbh, kbh, vth);
  k_ckl<<<dim3(ROWS), dim3(64), 0, stream>>>(q, kc, Wck, bck, ckl);

  // 4) dynamic 9-tap conv -> out[:, :, 384:768]
  k_convout<<<dim3(ROWS, NH), dim3(64), 0, stream>>>(co, ckl, out);

  // 5) flash attention -> out[:, :, 0:384]
  k_attn<<<dim3(SEQ / 128, BATCH * NH), dim3(256), 0, stream>>>(qbh, kbh, vth, out);
}
